// EquiMessageBlock_34376918237212
// MI455X (gfx1250) — matrix operand order repaired and checked by reading, not compiled, not run on hardware
//
#include <hip/hip_runtime.h>
#include <hip/hip_bf16.h>
#include <stdint.h>

#define FEAT    128
#define N_RBF   20
#define N_NODES 10000
#define N_EDGES 320000
#define THREEF  384
#define PI_F    3.14159265358979f

typedef __attribute__((ext_vector_type(16))) __bf16 v16bf;
typedef __attribute__((ext_vector_type(8)))  float  v8f;

__device__ inline v8f wmma_bf16(v16bf a, v16bf b, v8f c) {
  // D = A(16x32 bf16) * B(32x16 bf16) + C(16x16 f32)
  return __builtin_amdgcn_wmma_f32_16x16x32_bf16(false, a, false, b, (short)0, c, false, false);
}

// Fast 1-ulp reciprocal / sqrt (v_rcp_f32 / v_sqrt_f32, no IEEE div fixup chain)
__device__ inline float frcp(float x)  { return __builtin_amdgcn_rcpf(x); }
__device__ inline float fsqrt(float x) { return __builtin_amdgcn_sqrtf(x); }

// A-fragment (16x32 bf16) from row-major LDS tile, contiguous k order (matches prep_weights B layout):
// lane holds row M=lane&15; elems 0..15 -> K = kbase+hi*16+0..15
__device__ inline v16bf load_a_lds(const __bf16* A, int lda, int m, int kbase, int hi) {
  v16bf a;
  const __bf16* p = A + m * lda + kbase + hi * 16;
#pragma unroll
  for (int e = 0; e < 8; ++e) a[e] = p[e];
#pragma unroll
  for (int e = 0; e < 8; ++e) a[8 + e] = p[8 + e];
  return a;
}

// B-fragment from swizzled weights: 16 contiguous bf16 per (tile,lane) = one 32B load
__device__ inline v16bf load_b_sw(const __bf16* Wsw, int tileIdx, int lane) {
  const v16bf* p = (const v16bf*)(Wsw + ((size_t)tileIdx * 32 + lane) * 16);
  return *p;
}

// ---------------------------------------------------------------------------
// Kernel 0: convert fp32 weights -> bf16 in fragment-swizzled layout.
// Swizzled elem index i = (((kt*NT + nt)*32 + lane)*16 + j)
// maps to source (K = kt*32 + (lane>>4)*16 + j, col = nt*16 + (lane&15)).
// Wd is zero-padded from K=20 to K=32.
// ---------------------------------------------------------------------------
__global__ void prep_weights(const float* __restrict__ W1, const float* __restrict__ W2,
                             const float* __restrict__ Wd,
                             __bf16* __restrict__ W1s, __bf16* __restrict__ W2s,
                             __bf16* __restrict__ Wds) {
  int i = blockIdx.x * blockDim.x + threadIdx.x;
  int j = i & 15, lane = (i >> 4) & 31, t = i >> 9;
  int hi = lane >> 4, l = lane & 15;
  if (i < 4 * 8 * 512) {  // W1: 128x128, KT=4, NT=8
    int nt = t & 7, kt = t >> 3;
    int K = kt * 32 + hi * 16 + j, col = nt * 16 + l;
    W1s[i] = (__bf16)W1[K * FEAT + col];
  }
  if (i < 4 * 24 * 512) { // W2: 128x384, KT=4, NT=24
    int nt = t % 24, kt = t / 24;
    int K = kt * 32 + hi * 16 + j, col = nt * 16 + l;
    W2s[i] = (__bf16)W2[K * THREEF + col];
  }
  if (i < 1 * 24 * 512) { // Wd: 20x384 padded to 32x384, KT=1, NT=24
    int nt = t % 24;
    int K = hi * 16 + j, col = nt * 16 + l;
    Wds[i] = (K < N_RBF) ? (__bf16)Wd[K * THREEF + col] : (__bf16)0.0f;
  }
}

__global__ void zero_f32v4(float4* __restrict__ p, unsigned n4) {
  unsigned i = blockIdx.x * blockDim.x + threadIdx.x;
  if (i < n4) p[i] = make_float4(0.f, 0.f, 0.f, 0.f);
}

// ---------------------------------------------------------------------------
// Kernel 1: fused node GEMMs. Block = 16 rows, 256 threads (8 waves).
//   stage1: s tile -> LDS bf16
//   stage2: each wave one 16x16 tile of h = silu(s@W1+b1) -> LDS bf16
//   stage3: each wave three 16x16 tiles of phi = h@W2+b2 -> global f32
// ---------------------------------------------------------------------------
__global__ __launch_bounds__(256) void node_gemm(const float* __restrict__ s_j,
                                                 const float* __restrict__ b1,
                                                 const float* __restrict__ b2,
                                                 const __bf16* __restrict__ W1s,
                                                 const __bf16* __restrict__ W2s,
                                                 float* __restrict__ phi) {
  __shared__ __bf16 As[16 * 128];
  __shared__ __bf16 Hs[16 * 128];
  const int tid = threadIdx.x, lane = tid & 31, wave = tid >> 5;
  const int hi = lane >> 4, l = lane & 15;
  const int rowBase = blockIdx.x * 16;

  const float* srow = s_j + (size_t)rowBase * FEAT;
#pragma unroll
  for (int k = 0; k < 8; ++k) { int idx = tid + k * 256; As[idx] = (__bf16)srow[idx]; }
  __syncthreads();

  { // stage 2: h tile, cols [wave*16, wave*16+16)
    const int n0 = wave * 16;
    v8f c; float bv = b1[n0 + l];
#pragma unroll
    for (int r = 0; r < 8; ++r) c[r] = bv;
#pragma unroll
    for (int kt = 0; kt < 4; ++kt) {
      v16bf a = load_a_lds(As, 128, l, kt * 32, hi);
      v16bf b = load_b_sw(W1s, kt * 8 + wave, lane);
      c = wmma_bf16(a, b, c);
    }
#pragma unroll
    for (int r = 0; r < 8; ++r) {
      float x = c[r];
      float s = x * frcp(1.0f + __expf(-x));    // silu via v_rcp_f32 (no IEEE div chain)
      Hs[(r + hi * 8) * 128 + n0 + l] = (__bf16)s;
    }
  }
  __syncthreads();

  // stage 3: phi tiles nt = wave, wave+8, wave+16
#pragma unroll
  for (int t = 0; t < 3; ++t) {
    const int nt = wave + t * 8;
    const int n0 = nt * 16;
    v8f c; float bv = b2[n0 + l];
#pragma unroll
    for (int r = 0; r < 8; ++r) c[r] = bv;
#pragma unroll
    for (int kt = 0; kt < 4; ++kt) {
      v16bf a = load_a_lds(Hs, 128, l, kt * 32, hi);
      v16bf b = load_b_sw(W2s, kt * 24 + nt, lane);
      c = wmma_bf16(a, b, c);
    }
#pragma unroll
    for (int r = 0; r < 8; ++r)
      phi[(size_t)(rowBase + r + hi * 8) * THREEF + n0 + l] = c[r];
  }
}

// ---------------------------------------------------------------------------
// Kernel 2: edge kernel. Block = 128 edges (8 waves x 16-edge WMMA tiles).
// w_s = (rbf@Wd + bd) * f_cut via WMMA; epilogue gathers phi/v_j and
// scatter-adds delta_s / delta_v with f32 global atomics (L2-resident).
// Split 0/1/2 boundary is tile-uniform: part = nt >> 3.
// ---------------------------------------------------------------------------
__global__ __launch_bounds__(256) void edge_kernel(const float* __restrict__ r_ij,
                                                   const int64_t* __restrict__ nbrs,
                                                   const float* __restrict__ v_j,
                                                   const float* __restrict__ bd,
                                                   const __bf16* __restrict__ Wds,
                                                   const float* __restrict__ phi,
                                                   float* __restrict__ ds_out,
                                                   float* __restrict__ dv_out) {
  __shared__ int   dstL[128];
  __shared__ int   srcL[128];
  __shared__ float distL[128];
  __shared__ float fcutL[128];
  __shared__ float unitL[128][3];

  const int tid = threadIdx.x;
  const int eBase = blockIdx.x * 128;
  if (tid < 128) {
    int e = eBase + tid;
    dstL[tid] = (int)nbrs[2 * e];
    srcL[tid] = (int)nbrs[2 * e + 1];
    float x = r_ij[3 * e], y = r_ij[3 * e + 1], z = r_ij[3 * e + 2];
    float d = fsqrt(x * x + y * y + z * z + 3e-8f);
    float inv = frcp(d);
    distL[tid] = d;
    unitL[tid][0] = x * inv; unitL[tid][1] = y * inv; unitL[tid][2] = z * inv;
    fcutL[tid] = 0.5f * (__cosf(PI_F * d * 0.2f) + 1.0f) * (d < 5.0f ? 1.0f : 0.0f);
  }
  __syncthreads();

  const int lane = tid & 31, wave = tid >> 5, hi = lane >> 4, l = lane & 15;
  const int base16 = wave * 16;

  // A fragment: rbf matrix 16 edges x 32 (K=20 real, rest zero), computed in-register.
  const float d = distL[base16 + l];
  const float invd = frcp(d);
  v16bf a;
#pragma unroll
  for (int e2 = 0; e2 < 16; ++e2) {
    int K = hi * 16 + e2;
    float v = 0.0f;
    if (K < N_RBF) v = __sinf((float)(K + 1) * (PI_F * 0.2f) * d) * invd;
    a[e2] = (__bf16)v;
  }

#pragma unroll 2
  for (int nt = 0; nt < 24; ++nt) {
    const int col = nt * 16 + l;
    v16bf b = load_b_sw(Wds, nt, lane);
    v8f c; float bv = bd[col];
#pragma unroll
    for (int r = 0; r < 8; ++r) c[r] = bv;
    c = wmma_bf16(a, b, c);

    const int part = nt >> 3;     // 0: v_j term, 1: delta_s, 2: unit term
    const int f = col & 127;
#pragma unroll
    for (int r = 0; r < 8; ++r) {
      const int el = base16 + r + hi * 8;       // edge within block
      const float ws  = c[r] * fcutL[el];
      const int dst = dstL[el], src = srcL[el];
      const float ph  = phi[(size_t)src * THREEF + col];
      const float msg = ph * ws;
      if (part == 1) {
        atomicAdd(ds_out + (size_t)dst * FEAT + f, msg);
      } else if (part == 0) {
        const float* vj = v_j + ((size_t)src * FEAT + f) * 3;
        float* dv = dv_out + ((size_t)dst * FEAT + f) * 3;
        atomicAdd(dv + 0, msg * vj[0]);
        atomicAdd(dv + 1, msg * vj[1]);
        atomicAdd(dv + 2, msg * vj[2]);
      } else {
        float* dv = dv_out + ((size_t)dst * FEAT + f) * 3;
        atomicAdd(dv + 0, msg * unitL[el][0]);
        atomicAdd(dv + 1, msg * unitL[el][1]);
        atomicAdd(dv + 2, msg * unitL[el][2]);
      }
    }
  }
}

extern "C" void kernel_launch(void* const* d_in, const int* in_sizes, int n_in,
                              void* d_out, int out_size, void* d_ws, size_t ws_size,
                              hipStream_t stream) {
  (void)in_sizes; (void)n_in; (void)out_size; (void)ws_size;
  const float*   s_j  = (const float*)d_in[0];
  const float*   v_j  = (const float*)d_in[1];
  const float*   r_ij = (const float*)d_in[2];
  const int64_t* nbrs = (const int64_t*)d_in[3];
  const float*   W1   = (const float*)d_in[4];
  const float*   b1   = (const float*)d_in[5];
  const float*   W2   = (const float*)d_in[6];
  const float*   b2   = (const float*)d_in[7];
  const float*   Wd   = (const float*)d_in[8];
  const float*   bd   = (const float*)d_in[9];

  char* ws = (char*)d_ws;
  __bf16* W1s = (__bf16*)ws;                          // 16384 bf16 = 32768 B
  __bf16* W2s = (__bf16*)(ws + 32768);                // 49152 bf16 = 98304 B
  __bf16* Wds = (__bf16*)(ws + 32768 + 98304);        // 12288 bf16 = 24576 B
  float*  phi = (float*)(ws + 32768 + 98304 + 24576); // 10000*384 f32 = 15.36 MB

  float* ds_out = (float*)d_out;                      // delta_s_i: 10000*128
  float* dv_out = ds_out + (size_t)N_NODES * FEAT;    // delta_v_i: 10000*128*3

  hipLaunchKernelGGL(prep_weights, dim3(192), dim3(256), 0, stream, W1, W2, Wd, W1s, W2s, Wds);
  const unsigned n4 = N_NODES * FEAT;                 // 5,120,000 floats / 4
  hipLaunchKernelGGL(zero_f32v4, dim3((n4 + 255) / 256), dim3(256), 0, stream,
                     (float4*)d_out, n4);
  hipLaunchKernelGGL(node_gemm, dim3(N_NODES / 16), dim3(256), 0, stream,
                     s_j, b1, b2, W1s, W2s, phi);
  hipLaunchKernelGGL(edge_kernel, dim3(N_EDGES / 128), dim3(256), 0, stream,
                     r_ij, nbrs, v_j, bd, Wds, phi, ds_out, dv_out);
}
